// GConv_29317446763192
// MI455X (gfx1250) — matrix operand order repaired and checked by reading, not compiled, not run on hardware
//
#include <hip/hip_runtime.h>
#include <hip/hip_bf16.h>

typedef __attribute__((ext_vector_type(16))) _Float16 v16h;
typedef __attribute__((ext_vector_type(8)))  float    v8f;

// ---------------- conversion kernels ----------------

__global__ __launch_bounds__(256) void f32_to_f16_ker(const float* __restrict__ x,
                                                      _Float16* __restrict__ y, int n) {
  for (int i = blockIdx.x * blockDim.x + threadIdx.x; i < n; i += gridDim.x * blockDim.x)
    y[i] = (_Float16)x[i];
}

// y[n*K + k] = (f16) x[k*N + n]   (row-major [K][N] -> transposed [N][K])
__global__ __launch_bounds__(256) void f32_to_f16_T_ker(const float* __restrict__ x,
                                                        _Float16* __restrict__ y, int K, int N) {
  int i = blockIdx.x * blockDim.x + threadIdx.x;
  if (i >= K * N) return;
  int k = i / N, n = i - k * N;
  y[(size_t)n * K + k] = (_Float16)x[i];
}

// ---------------- reduction / normalize ----------------

__global__ __launch_bounds__(256) void sumsq_ker(const float* __restrict__ x, int n,
                                                 float* __restrict__ out) {
  __shared__ float sm[256];
  float s = 0.f;
  for (int i = blockIdx.x * blockDim.x + threadIdx.x; i < n; i += gridDim.x * blockDim.x) {
    float v = x[i];
    s += v * v;
  }
  sm[threadIdx.x] = s;
  __syncthreads();
  for (int off = 128; off > 0; off >>= 1) {
    if (threadIdx.x < (unsigned)off) sm[threadIdx.x] += sm[threadIdx.x + off];
    __syncthreads();
  }
  if (threadIdx.x == 0) atomicAdd(out, sm[0]);
}

__global__ __launch_bounds__(256) void scale_to_f16_ker(const float* __restrict__ x,
                                                        const float* __restrict__ sumsq,
                                                        _Float16* __restrict__ y, int n) {
  float sc = rsqrtf(*sumsq);  // x / ||x||_F
  for (int i = blockIdx.x * blockDim.x + threadIdx.x; i < n; i += gridDim.x * blockDim.x)
    y[i] = (_Float16)(x[i] * sc);
}

// ---------------- WMMA GEMM ----------------
// Block tile: 256(M) x 64(N), K-step 32, 256 threads = 8 waves (wave32).
// Each wave computes 32(M) x 64(N): 2 A-fragments x 4 B-fragments = 8 WMMAs/step.
// Tiles staged into double-buffered LDS with GLOBAL_LOAD_ASYNC_TO_LDS_B128
// (ASYNCcnt-tracked; INST_OFFSET applies to both LDS and global address).
// A staged row-major [256][32] halves; B column-major [64][32] halves from
// pre-transposed weights BT[N][K].
// EPI: 0 = relu -> f16 store, 1 = relu -> f32 store, 2 = graph scatter epilogue.

template <bool GATHER, int EPI>
__global__ __launch_bounds__(256) void gemm_wmma(
    const _Float16* __restrict__ A,       // [M][K]     (when !GATHER)
    const _Float16* __restrict__ obj16,   // gather sources (GATHER)
    const _Float16* __restrict__ pred16,
    const int* __restrict__ edges, int eS,
    const _Float16* __restrict__ BT,      // [N][K]
    const float* __restrict__ bias,       // [N]
    _Float16* __restrict__ out16,         // EPI 0
    float* __restrict__ outf,             // EPI 1; EPI 2: new_p destination
    float* __restrict__ pooled,           // EPI 2
    int N, int K) {
  __shared__ __align__(16) _Float16 As[2][256 * 32];
  __shared__ __align__(16) _Float16 Bs[2][64 * 32];

  const int t = threadIdx.x;
  const int wave = t >> 5;
  const int lane = t & 31;
  const int n0 = blockIdx.x * 64;
  const int m0 = blockIdx.y * 256;
  const int g = lane >> 4;    // which half of the wave
  const int ml = lane & 15;

  // ---- per-thread copy row (one thread = one 64B A-tile row)
  const _Float16* arow0 = nullptr;
  const _Float16* arow1 = nullptr;
  const _Float16* arow2 = nullptr;
  {
    int m = m0 + t;
    if (GATHER) {
      int b = m >> 13;          // T = 8192
      int s = edges[(size_t)m * 2 * eS];
      int o = edges[(size_t)m * 2 * eS + eS];
      arow0 = obj16 + ((size_t)((b << 10) + s)) * 384;   // subj
      arow1 = pred16 + (size_t)m * 384;                  // pred
      arow2 = obj16 + ((size_t)((b << 10) + o)) * 384;   // objt
    } else {
      arow0 = A + (size_t)m * K;
    }
  }
  // LDS byte offsets (low 32 bits of the generic shared pointer)
  const unsigned aLds[2] = {(unsigned)(size_t)(&As[0][t * 32]),
                            (unsigned)(size_t)(&As[1][t * 32])};
  // B chunk: col = t>>2, 16B chunk = t&3  -> dest halves at exactly t*8
  const unsigned bLds[2] = {(unsigned)(size_t)(&Bs[0][t * 8]),
                            (unsigned)(size_t)(&Bs[1][t * 8])};
  const int bCol = t >> 2, bCh = t & 3;

  v8f acc[8];
  {
    v8f z = {};
#pragma unroll
    for (int i = 0; i < 8; ++i) acc[i] = z;
  }

  // issue one K-step's async loads (5 instructions per thread)
  auto issueStage = [&](int buf, const _Float16* aseg, int kRel, int kAbs) {
    unsigned long long ag = (unsigned long long)(size_t)(aseg + kRel);
    unsigned long long bg =
        (unsigned long long)(size_t)(BT + (size_t)(n0 + bCol) * K + kAbs + bCh * 8);
    asm volatile(
        "global_load_async_to_lds_b128 %0, %2, off\n\t"
        "global_load_async_to_lds_b128 %0, %2, off offset:16\n\t"
        "global_load_async_to_lds_b128 %0, %2, off offset:32\n\t"
        "global_load_async_to_lds_b128 %0, %2, off offset:48\n\t"
        "global_load_async_to_lds_b128 %1, %3, off"
        :: "v"(aLds[buf]), "v"(bLds[buf]), "v"(ag), "v"(bg)
        : "memory");
  };

  auto compute = [&](int buf) {
    const _Float16* Ab = As[buf];
    const _Float16* Bb = Bs[buf];
    // A fragments (16x32): lane (g,ml) -> row ml, Ks {g*8+0..7} and {16+g*8+0..7}
    v16h a0, a1;
    {
      const uint4* r0 = (const uint4*)(Ab + (wave * 32 + ml) * 32);
      const uint4* r1 = (const uint4*)(Ab + (wave * 32 + 16 + ml) * 32);
      ((uint4*)&a0)[0] = r0[g];
      ((uint4*)&a0)[1] = r0[2 + g];
      ((uint4*)&a1)[0] = r1[g];
      ((uint4*)&a1)[1] = r1[2 + g];
    }
#pragma unroll
    for (int nt = 0; nt < 4; ++nt) {
      // B fragment (32x16): lane (g,ml) -> col nt*16+ml, Ks {g*8+0..7} and {16+g*8+0..7}
      v16h bf;
      const uint4* c4 = (const uint4*)(Bb + (nt * 16 + ml) * 32);
      ((uint4*)&bf)[0] = c4[g];
      ((uint4*)&bf)[1] = c4[2 + g];
      acc[nt] = __builtin_amdgcn_wmma_f32_16x16x32_f16(
          false, a0, false, bf, (short)0, acc[nt], false, false);
      acc[4 + nt] = __builtin_amdgcn_wmma_f32_16x16x32_f16(
          false, a1, false, bf, (short)0, acc[4 + nt], false, false);
    }
  };

  // double-buffered pipeline over one contiguous K segment (uniform control flow)
  auto runSegment = [&](const _Float16* aseg, int kAbs0, int kLen) {
    const int nsteps = kLen >> 5;
    issueStage(0, aseg, 0, kAbs0);
    for (int s = 0; s < nsteps; ++s) {
      int cur = s & 1;
      if (s + 1 < nsteps) {
        issueStage(cur ^ 1, aseg, (s + 1) * 32, kAbs0 + (s + 1) * 32);
        // 5 prefetch ops just issued; async loads complete in order, so
        // ASYNCcnt <= 5 means the current buffer's 5 ops have landed.
        asm volatile("s_wait_asynccnt 0x5" ::: "memory");
      } else {
        asm volatile("s_wait_asynccnt 0x0" ::: "memory");
      }
      __syncthreads();
      compute(cur);
      __syncthreads();
    }
  };

  if (GATHER) {  // K-steps never straddle the 384 boundaries: 3 uniform segments
    runSegment(arow0, 0, 384);
    runSegment(arow1, 384, 384);
    runSegment(arow2, 768, 384);
  } else {
    runSegment(arow0, 0, K);
  }

  // ---- epilogue. acc[mi*4+nt][j] -> M = wave*32 + mi*16 + g*8 + j, N = nt*16 + ml
#pragma unroll
  for (int mi = 0; mi < 2; ++mi) {
    const int mbase = m0 + wave * 32 + mi * 16 + g * 8;
    int sA[8], oA[8];
    if (EPI == 2) {
#pragma unroll
      for (int j = 0; j < 8; ++j) {
        int mj = mbase + j;
        sA[j] = edges[(size_t)mj * 2 * eS];
        oA[j] = edges[(size_t)mj * 2 * eS + eS];
      }
    }
#pragma unroll
    for (int nt = 0; nt < 4; ++nt) {
      int n = n0 + nt * 16 + ml;
      float bb = bias[n];
#pragma unroll
      for (int j = 0; j < 8; ++j) {
        float v = acc[mi * 4 + nt][j] + bb;
        v = v > 0.f ? v : 0.f;
        int mj = mbase + j;
        if (EPI == 0) {
          out16[(size_t)mj * N + n] = (_Float16)v;
        } else if (EPI == 1) {
          outf[(size_t)mj * N + n] = v;
        } else {
          int b = mj >> 13;
          if (n < 384)
            atomicAdd(&pooled[(size_t)((b << 10) + sA[j]) * 384 + n], v);         // new_s
          else if (n < 768)
            outf[(size_t)mj * 384 + (n - 384)] = v;                                // new_p
          else
            atomicAdd(&pooled[(size_t)((b << 10) + oA[j]) * 384 + (n - 768)], v);  // new_o
        }
      }
    }
  }
}

// ---------------- launcher ----------------

extern "C" void kernel_launch(void* const* d_in, const int* in_sizes, int n_in,
                              void* d_out, int out_size, void* d_ws, size_t ws_size,
                              hipStream_t stream) {
  const float* obj_f   = (const float*)d_in[0];   // (8,1024,384)
  const float* pred_f  = (const float*)d_in[1];   // (8,8192,384)
  const int*   edges   = (const int*)d_in[2];     // (8,8192,2)
  const float* w1a     = (const float*)d_in[3];   // (1152,384)
  const float* b1a     = (const float*)d_in[4];
  const float* w1b     = (const float*)d_in[5];   // (384,1152)
  const float* b1b     = (const float*)d_in[6];
  const float* w2a     = (const float*)d_in[7];   // (384,384)
  const float* b2a     = (const float*)d_in[8];
  const float* w2b     = (const float*)d_in[9];   // (384,384)
  const float* b2b     = (const float*)d_in[10];

  const int OBJ_N  = 8 * 1024 * 384;   // 3,145,728
  const int PRED_N = 8 * 8192 * 384;   // 25,165,824
  const int M1     = 8 * 8192;         // 65,536 rows
  const int MO     = 8 * 1024;         // 8,192 rows
  // int64 vs int32 edges hedge: if the harness hands 64-bit ints viewed as
  // int32 pairs (little-endian, values < 1024 so high word is 0), stride 2.
  const int eS = (in_sizes[2] >= 8 * 8192 * 2 * 2) ? 2 : 1;

  char* ws = (char*)d_ws;
  size_t off = 0;
  auto alloc = [&](size_t bytes) {
    size_t p = off;
    off = (off + bytes + 255) & ~(size_t)255;
    return p;
  };
  size_t o_obj16   = alloc((size_t)OBJ_N * 2);
  size_t o_pred16  = alloc((size_t)PRED_N * 2);
  size_t o_w1aT    = alloc((size_t)1152 * 384 * 2);
  size_t o_w1bT    = alloc((size_t)384 * 1152 * 2);
  size_t o_w2aT    = alloc((size_t)384 * 384 * 2);
  size_t o_w2bT    = alloc((size_t)384 * 384 * 2);
  size_t o_h16     = alloc((size_t)M1 * 384 * 2);
  size_t o_pooled  = alloc((size_t)OBJ_N * 4);
  size_t o_sumsq   = alloc(256);
  size_t o_pooln16 = alloc((size_t)OBJ_N * 2);
  size_t o_hid216  = alloc((size_t)OBJ_N * 2);

  _Float16* obj16   = (_Float16*)(ws + o_obj16);
  _Float16* pred16  = (_Float16*)(ws + o_pred16);
  _Float16* w1aT    = (_Float16*)(ws + o_w1aT);
  _Float16* w1bT    = (_Float16*)(ws + o_w1bT);
  _Float16* w2aT    = (_Float16*)(ws + o_w2aT);
  _Float16* w2bT    = (_Float16*)(ws + o_w2bT);
  _Float16* h16     = (_Float16*)(ws + o_h16);
  float*    pooled  = (float*)(ws + o_pooled);
  float*    sumsq   = (float*)(ws + o_sumsq);
  _Float16* pooln16 = (_Float16*)(ws + o_pooln16);
  _Float16* hid216  = (_Float16*)(ws + o_hid216);

  float* out_obj = (float*)d_out;                 // (8,1024,384)
  float* out_p   = (float*)d_out + OBJ_N;         // (8,8192,384)

  // zero pooled accumulator + sumsq (contiguous in ws)
  hipMemsetAsync(ws + o_pooled, 0, (o_sumsq - o_pooled) + 256, stream);

  // f32 -> f16 conversions
  f32_to_f16_ker<<<1024, 256, 0, stream>>>(obj_f, obj16, OBJ_N);
  f32_to_f16_ker<<<2048, 256, 0, stream>>>(pred_f, pred16, PRED_N);
  f32_to_f16_T_ker<<<(1152 * 384 + 255) / 256, 256, 0, stream>>>(w1a, w1aT, 1152, 384);
  f32_to_f16_T_ker<<<(384 * 1152 + 255) / 256, 256, 0, stream>>>(w1b, w1bT, 384, 1152);
  f32_to_f16_T_ker<<<(384 * 384 + 255) / 256, 256, 0, stream>>>(w2a, w2aT, 384, 384);
  f32_to_f16_T_ker<<<(384 * 384 + 255) / 256, 256, 0, stream>>>(w2b, w2bT, 384, 384);

  // GEMM1 (gathered triple @ w1a, relu) -> h16 : M=65536, N=384, K=1152
  gemm_wmma<true, 0><<<dim3(384 / 64, M1 / 256), 256, 0, stream>>>(
      nullptr, obj16, pred16, edges, eS, w1aT, b1a, h16, nullptr, nullptr, 384, 1152);

  // GEMM2 (h @ w1b, relu) -> scatter new_s/new_o into pooled, new_p to d_out
  gemm_wmma<false, 2><<<dim3(1152 / 64, M1 / 256), 256, 0, stream>>>(
      h16, nullptr, nullptr, edges, eS, w1bT, b1b, nullptr, out_p, pooled, 1152, 384);

  // Frobenius norm of pooled, then pooled/||pooled|| -> f16
  sumsq_ker<<<512, 256, 0, stream>>>(pooled, OBJ_N, sumsq);
  scale_to_f16_ker<<<1024, 256, 0, stream>>>(pooled, sumsq, pooln16, OBJ_N);

  // GEMM3 (pooled_n @ w2a, relu) -> hid2 : M=8192, N=384, K=384
  gemm_wmma<false, 0><<<dim3(384 / 64, MO / 256), 256, 0, stream>>>(
      pooln16, nullptr, nullptr, edges, eS, w2aT, b2a, hid216, nullptr, nullptr, 384, 384);

  // GEMM4 (hid2 @ w2b, relu) -> out_obj (f32)
  gemm_wmma<false, 1><<<dim3(384 / 64, MO / 256), 256, 0, stream>>>(
      hid216, nullptr, nullptr, edges, eS, w2bT, b2b, nullptr, out_obj, nullptr, 384, 384);

  (void)n_in; (void)out_size; (void)ws_size;
}
